// _AltBlock_28707561406477
// MI455X (gfx1250) — matrix operand order repaired and checked by reading, not compiled, not run on hardware
//
#include <hip/hip_runtime.h>
#include <hip/hip_bf16.h>
#include <math.h>

// ---------------------------------------------------------------- constants
#define kB   2
#define kN   2048
#define kC   768
#define kH   12
#define kD   64
#define kHID 3072
#define kBNC ((size_t)kB * kN * kC)   // 3,145,728

typedef __attribute__((ext_vector_type(16))) _Float16 v16h;
typedef __attribute__((ext_vector_type(8)))  _Float16 v8h;
typedef __attribute__((ext_vector_type(8)))  float    v8f;

// ---------------------------------------------------------------- fragments
// A fragment: 16x32 (MxK), row-major source with leading dim lda (halves).
// Per ISA 7.12.2: lane L holds row M=L&15; lanes 0-15 carry K=0..7 & 16..23,
// lanes 16-31 carry K=8..15 & 24..31.
__device__ __forceinline__ v16h load_frag_a(const _Float16* base, int lda) {
  const int lane = threadIdx.x & 31;
  const int m  = lane & 15;
  const int ks = (lane >> 4) * 8;
  const _Float16* p = base + (size_t)m * lda + ks;
  v8h lo = *(const v8h*)(p);
  v8h hi = *(const v8h*)(p + 16);
  v16h r;
#pragma unroll
  for (int i = 0; i < 8; ++i) { r[i] = lo[i]; r[i + 8] = hi[i]; }
  return r;
}

// B fragment: 32x16 (KxN) where B[k][n] = W[n][k], W row-major [N,K] (ldw).
// Lane L holds column N=L&15; lanes 0-15 carry K=0..7 & 16..23,
// lanes 16-31 carry K=8..15 & 24..31.
__device__ __forceinline__ v16h load_frag_b(const _Float16* base, int ldw) {
  const int lane = threadIdx.x & 31;
  const int n  = lane & 15;
  const int kh = (lane >> 4) * 8;
  const _Float16* p = base + (size_t)n * ldw + kh;
  v8h lo = *(const v8h*)(p);
  v8h hi = *(const v8h*)(p + 16);
  v16h r;
#pragma unroll
  for (int i = 0; i < 8; ++i) { r[i] = lo[i]; r[i + 8] = hi[i]; }
  return r;
}

__device__ __forceinline__ v8f wmma_f16(v16h a, v16h b, v8f c) {
  return __builtin_amdgcn_wmma_f32_16x16x32_f16(
      /*neg_a=*/false, a, /*neg_b=*/false, b,
      /*c_mod=*/(short)0, c, /*reuse_a=*/false, /*reuse_b=*/false);
}

__device__ __forceinline__ float gelu_exact(float x) {
  return 0.5f * x * (1.0f + erff(x * 0.70710678118654752f));
}

// ---------------------------------------------------------------- f32 -> f16
__global__ void cvt_f32_to_f16(const float* __restrict__ in,
                               _Float16* __restrict__ out, int n) {
  int i = blockIdx.x * blockDim.x + threadIdx.x;
  if (i < n) out[i] = (_Float16)in[i];
}

// ---------------------------------------------------------------- layernorm
// One block per row of C=768; output f16 for the downstream WMMA GEMM.
__global__ __launch_bounds__(256) void layernorm_f16(
    const float* __restrict__ x, const float* __restrict__ w,
    const float* __restrict__ bs, _Float16* __restrict__ out) {
  const int row = blockIdx.x;
  const int tid = threadIdx.x;
  const float* xr = x + (size_t)row * kC;
  __shared__ float sh[512];
  float s = 0.f, s2 = 0.f;
  for (int i = tid; i < kC; i += 256) { float v = xr[i]; s += v; s2 += v * v; }
  sh[tid] = s; sh[256 + tid] = s2;
  __syncthreads();
  for (int o = 128; o > 0; o >>= 1) {
    if (tid < o) { sh[tid] += sh[tid + o]; sh[256 + tid] += sh[256 + tid + o]; }
    __syncthreads();
  }
  const float mean = sh[0] * (1.0f / kC);
  const float var  = sh[256] * (1.0f / kC) - mean * mean;
  const float inv  = rsqrtf(var + 1e-5f);
  for (int i = tid; i < kC; i += 256)
    out[(size_t)row * kC + i] = (_Float16)((xr[i] - mean) * inv * w[i] + bs[i]);
}

// ---------------------------------------------------------------- WMMA GEMM
// Out[M,Nt] = epilogue( A[M,K] @ W[Nt,K]^T + bias )
// 128 threads = 4 waves; block tile 128x64; wave w owns rows [w*32, w*32+32).
// Per k-step each wave issues 8 WMMAs from 12 b128 loads, with explicit
// register double-buffering so loads for step k+1 overlap the 8 WMMAs of k.
__global__ __launch_bounds__(128) void gemm_wmma(
    const _Float16* __restrict__ A, const _Float16* __restrict__ W,
    const float* __restrict__ bias, const float* __restrict__ resid,
    float* __restrict__ out32, _Float16* __restrict__ out16,
    int M, int Nt, int K, int act, long dual_off) {
  const int wave = threadIdx.x >> 5;
  const int lane = threadIdx.x & 31;
  const int row0 = blockIdx.y * 128 + wave * 32;
  const int n0   = blockIdx.x * 64;

  const _Float16* A0 = A + (size_t)row0 * K;
  const _Float16* A1 = A + (size_t)(row0 + 16) * K;
  const _Float16* W0 = W + (size_t)(n0 +  0) * K;
  const _Float16* W1 = W + (size_t)(n0 + 16) * K;
  const _Float16* W2 = W + (size_t)(n0 + 32) * K;
  const _Float16* W3 = W + (size_t)(n0 + 48) * K;

  v8f c00 = {}, c01 = {}, c02 = {}, c03 = {};
  v8f c10 = {}, c11 = {}, c12 = {}, c13 = {};

  // prime the pipeline
  v16h a0 = load_frag_a(A0, K), a1 = load_frag_a(A1, K);
  v16h b0 = load_frag_b(W0, K), b1 = load_frag_b(W1, K);
  v16h b2 = load_frag_b(W2, K), b3 = load_frag_b(W3, K);

  for (int k = 32; k < K; k += 32) {
    // issue next tile's loads first; WMMAs below hide their latency
    v16h na0 = load_frag_a(A0 + k, K);
    v16h na1 = load_frag_a(A1 + k, K);
    v16h nb0 = load_frag_b(W0 + k, K);
    v16h nb1 = load_frag_b(W1 + k, K);
    v16h nb2 = load_frag_b(W2 + k, K);
    v16h nb3 = load_frag_b(W3 + k, K);
    if (k + 32 < K) {  // WGP-scope prefetch of the A stream one step ahead
      __builtin_prefetch(A0 + k + 32, 0, 3);
      __builtin_prefetch(A1 + k + 32, 0, 3);
    }
    c00 = wmma_f16(a0, b0, c00);
    c01 = wmma_f16(a0, b1, c01);
    c02 = wmma_f16(a0, b2, c02);
    c03 = wmma_f16(a0, b3, c03);
    c10 = wmma_f16(a1, b0, c10);
    c11 = wmma_f16(a1, b1, c11);
    c12 = wmma_f16(a1, b2, c12);
    c13 = wmma_f16(a1, b3, c13);
    a0 = na0; a1 = na1;
    b0 = nb0; b1 = nb1; b2 = nb2; b3 = nb3;
  }
  // drain
  c00 = wmma_f16(a0, b0, c00);
  c01 = wmma_f16(a0, b1, c01);
  c02 = wmma_f16(a0, b2, c02);
  c03 = wmma_f16(a0, b3, c03);
  c10 = wmma_f16(a1, b0, c10);
  c11 = wmma_f16(a1, b1, c11);
  c12 = wmma_f16(a1, b2, c12);
  c13 = wmma_f16(a1, b3, c13);

  const int half = lane >> 4, col = lane & 15;
#pragma unroll
  for (int rt = 0; rt < 2; ++rt) {
#pragma unroll
    for (int j = 0; j < 4; ++j) {
      v8f cc;
      if (rt == 0) cc = (j == 0) ? c00 : (j == 1) ? c01 : (j == 2) ? c02 : c03;
      else         cc = (j == 0) ? c10 : (j == 1) ? c11 : (j == 2) ? c12 : c13;
#pragma unroll
      for (int r = 0; r < 8; ++r) {
        const int row = row0 + rt * 16 + r + half * 8;
        const int nn  = n0 + j * 16 + col;
        float v = cc[r] + bias[nn];
        const size_t idx = (size_t)row * Nt + nn;
        if (resid) v += resid[idx];
        if (act == 1) v = gelu_exact(v);
        if (dual_off > 0) {            // final FC2: d_out = [2t | t]
          out32[idx] = 2.0f * v;
          out32[(size_t)dual_off + idx] = v;
        } else if (out32) {
          out32[idx] = v;
        }
        if (out16) out16[idx] = (_Float16)v;
      }
    }
  }
}

// ---------------------------------------------------------------- V^T build
// vt[b,h,d,n] = qkv_h[(b*N+n)*3C + 2C + h*D + d]  (f16)
__global__ void build_vt(const _Float16* __restrict__ qkv,
                         _Float16* __restrict__ vt) {
  size_t i = (size_t)blockIdx.x * 256 + threadIdx.x;
  const size_t total = (size_t)kB * kH * kD * kN;
  if (i >= total) return;
  int n = (int)(i % kN); size_t t = i / kN;
  int d = (int)(t % kD); t /= kD;
  int h = (int)(t % kH); int b = (int)(t / kH);
  vt[i] = qkv[((size_t)(b * kN + n)) * (3 * kC) + 2 * kC + h * kD + d];
}

// ---------------------------------------------------------------- attention
// Flash attention: 1 wave per (b, h, 16 query rows). QK^T and PV via WMMA,
// online softmax across the 16-lane halves, P re-laid-out through LDS.
// Bandwidth-bound on the 402MB alibi stream (~17us floor at 23.3 TB/s).
__global__ __launch_bounds__(32) void attn_flash(
    const _Float16* __restrict__ qkv, const _Float16* __restrict__ vt,
    const float* __restrict__ alibi, const unsigned char* __restrict__ mask,
    _Float16* __restrict__ attn_out) {
  __shared__ _Float16 Pb[16 * 32];
  const int lane = threadIdx.x;
  const int half = lane >> 4, col = lane & 15;
  const int n0 = blockIdx.x * 16;
  const int h  = blockIdx.y;
  const int b  = blockIdx.z;
  const int ld = 3 * kC;
  const float scale = 0.125f;  // D^-0.5 = 1/8

  const _Float16* qbase = qkv + ((size_t)(b * kN + n0)) * ld + h * kD;
  const v16h qa0 = load_frag_a(qbase, ld);
  const v16h qa1 = load_frag_a(qbase + 32, ld);

  float mrun[8], lrun[8];
  v8f acc0 = {}, acc1 = {}, acc2 = {}, acc3 = {};
#pragma unroll
  for (int r = 0; r < 8; ++r) { mrun[r] = -INFINITY; lrun[r] = 0.f; }

  const float* ab_row = alibi + (((size_t)(b * kH + h)) * kN + n0) * kN;
  const unsigned char* mrow = mask + (size_t)b * kN;
  const _Float16* vt_bh = vt + ((size_t)(b * kH + h)) * kD * kN;

  for (int m0 = 0; m0 < kN; m0 += 32) {
    // ---- scores: two 16x16 tiles, K=64 reduced by 2 WMMAs each
    v8f s0 = {}, s1 = {};
    {
      const _Float16* kb = qkv + ((size_t)(b * kN + m0)) * ld + kC + h * kD;
      v16h b0 = load_frag_b(kb, ld);
      v16h b1 = load_frag_b(kb + 32, ld);
      s0 = wmma_f16(qa0, b0, s0);
      s0 = wmma_f16(qa1, b1, s0);
      const _Float16* kb2 = kb + (size_t)16 * ld;
      v16h b2 = load_frag_b(kb2, ld);
      v16h b3 = load_frag_b(kb2 + 32, ld);
      s1 = wmma_f16(qa0, b2, s1);
      s1 = wmma_f16(qa1, b3, s1);
    }
    // ---- alibi + mask + online softmax (per C-fragment row layout)
#pragma unroll
    for (int r = 0; r < 8; ++r) {
      const int row = r + half * 8;
      const int c0 = m0 + col, c1 = m0 + 16 + col;
      float v0 = s0[r] * scale + ab_row[(size_t)row * kN + c0];
      float v1 = s1[r] * scale + ab_row[(size_t)row * kN + c1];
      if (mrow[c0]) v0 = -INFINITY;
      if (mrow[c1]) v1 = -INFINITY;
      float mx = fmaxf(v0, v1);
#pragma unroll
      for (int o = 1; o < 16; o <<= 1) mx = fmaxf(mx, __shfl_xor(mx, o, 32));
      const float mn   = fmaxf(mrun[r], mx);
      const float corr = __expf(mrun[r] - mn);
      const float p0 = __expf(v0 - mn);
      const float p1 = __expf(v1 - mn);
      float ps = p0 + p1;
#pragma unroll
      for (int o = 1; o < 16; o <<= 1) ps += __shfl_xor(ps, o, 32);
      lrun[r] = lrun[r] * corr + ps;
      mrun[r] = mn;
      acc0[r] *= corr; acc1[r] *= corr; acc2[r] *= corr; acc3[r] *= corr;
      Pb[row * 32 + col]      = (_Float16)p0;
      Pb[row * 32 + 16 + col] = (_Float16)p1;
    }
    __syncthreads();
    // ---- PV: P(16x32) @ V(32x64) via 4 WMMAs against pre-transposed V^T
    const v16h pa = load_frag_a(Pb, 32);
    const _Float16* vb = vt_bh + m0;
    v16h b0 = load_frag_b(vb + (size_t) 0 * kN, kN);
    v16h b1 = load_frag_b(vb + (size_t)16 * kN, kN);
    v16h b2 = load_frag_b(vb + (size_t)32 * kN, kN);
    v16h b3 = load_frag_b(vb + (size_t)48 * kN, kN);
    acc0 = wmma_f16(pa, b0, acc0);
    acc1 = wmma_f16(pa, b1, acc1);
    acc2 = wmma_f16(pa, b2, acc2);
    acc3 = wmma_f16(pa, b3, acc3);
    __syncthreads();
  }

  // ---- normalize and store [B,N,C] f16 for the proj GEMM
#pragma unroll
  for (int r = 0; r < 8; ++r) {
    const int row = n0 + r + half * 8;
    const float inv = 1.0f / lrun[r];
    _Float16* o = attn_out + ((size_t)(b * kN + row)) * kC + h * kD + col;
    o[0]  = (_Float16)(acc0[r] * inv);
    o[16] = (_Float16)(acc1[r] * inv);
    o[32] = (_Float16)(acc2[r] * inv);
    o[48] = (_Float16)(acc3[r] * inv);
  }
}

// ---------------------------------------------------------------- launcher
extern "C" void kernel_launch(void* const* d_in, const int* in_sizes, int n_in,
                              void* d_out, int out_size, void* d_ws, size_t ws_size,
                              hipStream_t stream) {
  (void)in_sizes; (void)n_in; (void)out_size; (void)ws_size;
  const float*         x       = (const float*)d_in[0];
  const unsigned char* pmask   = (const unsigned char*)d_in[1];  // jax bool
  const float*         alibi   = (const float*)d_in[2];
  const float*         qkv_w   = (const float*)d_in[3];
  const float*         qkv_b   = (const float*)d_in[4];
  const float*         proj_w  = (const float*)d_in[5];
  const float*         proj_b  = (const float*)d_in[6];
  const float*         n1_w    = (const float*)d_in[7];
  const float*         n1_b    = (const float*)d_in[8];
  const float*         n2_w    = (const float*)d_in[9];
  const float*         n2_b    = (const float*)d_in[10];
  const float*         fc1_w   = (const float*)d_in[11];
  const float*         fc1_b   = (const float*)d_in[12];
  const float*         fc2_w   = (const float*)d_in[13];
  const float*         fc2_b   = (const float*)d_in[14];
  float* out = (float*)d_out;

  // ---- workspace carve-up
  char* p = (char*)d_ws;
  float* h32 = (float*)p;              p += kBNC * sizeof(float);
  _Float16* qkv_w16 = (_Float16*)p;    p += (size_t)3 * kC * kC * 2;
  _Float16* proj_w16 = (_Float16*)p;   p += (size_t)kC * kC * 2;
  _Float16* fc1_w16 = (_Float16*)p;    p += (size_t)kHID * kC * 2;
  _Float16* fc2_w16 = (_Float16*)p;    p += (size_t)kC * kHID * 2;
  _Float16* xn_h   = (_Float16*)p;     p += kBNC * 2;
  _Float16* qkv_h  = (_Float16*)p;     p += (size_t)kB * kN * 3 * kC * 2;
  _Float16* vt_h   = (_Float16*)p;     p += (size_t)kB * kH * kD * kN * 2;
  _Float16* attn_h = (_Float16*)p;     p += kBNC * 2;
  _Float16* hn_h   = (_Float16*)p;     p += kBNC * 2;
  _Float16* g_h    = (_Float16*)p;     p += (size_t)kB * kN * kHID * 2;

  const int M = kB * kN;  // 4096

  // ---- 1. weights to f16 (all fit comfortably in 192MB L2)
  {
    int n;
    n = 3 * kC * kC;  cvt_f32_to_f16<<<(n + 255) / 256, 256, 0, stream>>>(qkv_w, qkv_w16, n);
    n = kC * kC;      cvt_f32_to_f16<<<(n + 255) / 256, 256, 0, stream>>>(proj_w, proj_w16, n);
    n = kHID * kC;    cvt_f32_to_f16<<<(n + 255) / 256, 256, 0, stream>>>(fc1_w, fc1_w16, n);
    n = kC * kHID;    cvt_f32_to_f16<<<(n + 255) / 256, 256, 0, stream>>>(fc2_w, fc2_w16, n);
  }

  // ---- 2. LN1 -> xn (f16)
  layernorm_f16<<<M, 256, 0, stream>>>(x, n1_w, n1_b, xn_h);

  // ---- 3. QKV GEMM: [4096,768] @ [2304,768]^T -> qkv_h (f16)
  gemm_wmma<<<dim3(3 * kC / 64, M / 128), 128, 0, stream>>>(
      xn_h, qkv_w16, qkv_b, nullptr, nullptr, qkv_h, M, 3 * kC, kC, 0, 0);

  // ---- 4. V^T for PV-matmul B-fragments
  {
    size_t total = (size_t)kB * kH * kD * kN;
    build_vt<<<(unsigned)((total + 255) / 256), 256, 0, stream>>>(qkv_h, vt_h);
  }

  // ---- 5. flash attention (alibi + padding mask)
  attn_flash<<<dim3(kN / 16, kH, kB), 32, 0, stream>>>(
      qkv_h, vt_h, alibi, pmask, attn_h);

  // ---- 6. proj GEMM + residual x -> h (f32)
  gemm_wmma<<<dim3(kC / 64, M / 128), 128, 0, stream>>>(
      attn_h, proj_w16, proj_b, x, h32, nullptr, M, kC, kC, 0, 0);

  // ---- 7. LN2 -> hn (f16)
  layernorm_f16<<<M, 256, 0, stream>>>(h32, n2_w, n2_b, hn_h);

  // ---- 8. FC1 GEMM + exact GELU -> g (f16)
  gemm_wmma<<<dim3(kHID / 64, M / 128), 128, 0, stream>>>(
      hn_h, fc1_w16, fc1_b, nullptr, nullptr, g_h, M, kHID, kC, 1, 0);

  // ---- 9. FC2 GEMM -> d_out = [2t | t]
  gemm_wmma<<<dim3(kC / 64, M / 128), 128, 0, stream>>>(
      g_h, fc2_w16, fc2_b, nullptr, out, nullptr, M, kC, kHID, 0, (long)kBNC);
}
